// GCN_29437705846971
// MI455X (gfx1250) — matrix operand order repaired and checked by reading, not compiled, not run on hardware
//
#include <hip/hip_runtime.h>
#include <hip/hip_bf16.h>

#define NNODES 100000
#define FDIM   64
#define NCLS   40

typedef __attribute__((ext_vector_type(16))) __bf16        v16bf;
typedef __attribute__((ext_vector_type(8)))  float         v8f;
typedef __attribute__((ext_vector_type(4)))  unsigned int  u32x4;

union FragBF16 {
    u32x4 u[2];   // two 16-byte vector loads
    v16bf v;      // WMMA operand view
};

// ---------------- degree / norm ----------------
__global__ void k_init_deg(float* deg, int n) {
    int i = blockIdx.x * blockDim.x + threadIdx.x;
    if (i < n) deg[i] = 1.0f;                      // self-loop
}

__global__ void k_count_deg(const int* __restrict__ dst, int e, float* deg) {
    int i = blockIdx.x * blockDim.x + threadIdx.x;
    if (i < e) unsafeAtomicAdd(&deg[dst[i]], 1.0f);
}

__global__ void k_dinv(float* deg, int n) {        // in-place deg -> deg^-1/2
    int i = blockIdx.x * blockDim.x + threadIdx.x;
    if (i < n) deg[i] = rsqrtf(deg[i]);
}

// ---------------- bf16 conversion ----------------
// two floats in, one packed dword (2x bf16) out
__global__ void k_cvt_x(const float2* __restrict__ x, unsigned int* __restrict__ xb, int n2) {
    int i = blockIdx.x * blockDim.x + threadIdx.x;
    if (i < n2) {
        float2 v = x[i];
        union { __bf16 h[2]; unsigned int u; } p;
        p.h[0] = (__bf16)v.x;
        p.h[1] = (__bf16)v.y;
        xb[i] = p.u;
    }
}

// W,R are (64, ncols) row-major; write TRANSPOSED + zero-padded to 64x64:
// wbT[n*64 + k] = W[k*ncols + n]  (so a B column is contiguous in K)
__global__ void k_cvt_w(const float* __restrict__ W, const float* __restrict__ R,
                        __bf16* __restrict__ wbT, __bf16* __restrict__ rbT, int ncols) {
    int i = blockIdx.x * blockDim.x + threadIdx.x;  // 0..4095
    if (i >= 64 * 64) return;
    int n = i >> 6, k = i & 63;
    wbT[i] = (n < ncols) ? (__bf16)W[k * ncols + n] : (__bf16)0.0f;
    rbT[i] = (n < ncols) ? (__bf16)R[k * ncols + n] : (__bf16)0.0f;
}

// ---------------- dual GEMM: h = xb@W, r = xb@R  (N x 64, K = 64) ----------------
__global__ __launch_bounds__(256) void k_gemm_dual(
    const __bf16* __restrict__ xb, const __bf16* __restrict__ wbT,
    const __bf16* __restrict__ rbT, float* __restrict__ h, float* __restrict__ r) {
    const int lane = threadIdx.x & 31;
    const int wave = (blockIdx.x * blockDim.x + threadIdx.x) >> 5;
    const int mt = wave >> 2;                       // 6250 row tiles (exact)
    const int nt = wave & 3;                        // 4 col tiles
    if (mt >= (NNODES >> 4)) return;
    const int half = lane >> 4;
    const int mrow = (mt << 4) + (lane & 15);       // A row for this lane
    const int ncol = (nt << 4) + (lane & 15);       // B/C/D column for this lane

    // A 16x32 bf16 (ISA 7.12.2): lane-half h holds K = 8h..8h+7 (elems 0..7)
    // and K = 16+8h..23+8h (elems 8..15) of its row -> two contiguous b128 loads.
    const __bf16* arow = xb + (size_t)mrow * FDIM;
    FragBF16 a_lo, a_hi;
    a_lo.u[0] = *(const u32x4*)(arow +      (half << 3));
    a_lo.u[1] = *(const u32x4*)(arow + 16 + (half << 3));
    a_hi.u[0] = *(const u32x4*)(arow + 32 + (half << 3));
    a_hi.u[1] = *(const u32x4*)(arow + 48 + (half << 3));

    // B 32x16 bf16: elems 0..7 = K 8h..8h+7, elems 8..15 = K 16+8h..23+8h (N = ncol);
    // with transposed weights -> two b128 loads per fragment.
    const __bf16* wc = wbT + (size_t)ncol * FDIM + (half << 3);
    const __bf16* rc = rbT + (size_t)ncol * FDIM + (half << 3);
    FragBF16 bw_lo, bw_hi, br_lo, br_hi;
    bw_lo.u[0] = *(const u32x4*)(wc);
    bw_lo.u[1] = *(const u32x4*)(wc + 16);
    bw_hi.u[0] = *(const u32x4*)(wc + 32);
    bw_hi.u[1] = *(const u32x4*)(wc + 48);
    br_lo.u[0] = *(const u32x4*)(rc);
    br_lo.u[1] = *(const u32x4*)(rc + 16);
    br_hi.u[0] = *(const u32x4*)(rc + 32);
    br_hi.u[1] = *(const u32x4*)(rc + 48);

    v8f ch = {}, cr = {};
    ch = __builtin_amdgcn_wmma_f32_16x16x32_bf16(false, a_lo.v, false, bw_lo.v, (short)0, ch, false, false);
    ch = __builtin_amdgcn_wmma_f32_16x16x32_bf16(false, a_hi.v, false, bw_hi.v, (short)0, ch, false, false);
    cr = __builtin_amdgcn_wmma_f32_16x16x32_bf16(false, a_lo.v, false, br_lo.v, (short)0, cr, false, false);
    cr = __builtin_amdgcn_wmma_f32_16x16x32_bf16(false, a_hi.v, false, br_hi.v, (short)0, cr, false, false);

    // C/D 16x16 f32: VGPR t -> row t + 8*half, column = ncol.
    #pragma unroll
    for (int t = 0; t < 8; ++t) {
        int m = (mt << 4) + t + (half << 3);
        h[(size_t)m * FDIM + ncol] = ch[t];
        r[(size_t)m * FDIM + ncol] = cr[t];
    }
}

// ---------------- aggregation ----------------
// self-loop init: agg[i,f] = dinv[i]^2 * h[i,f]
__global__ void k_self(const float* __restrict__ dinv, const float* __restrict__ h,
                       float* __restrict__ agg, int total) {
    int i = blockIdx.x * blockDim.x + threadIdx.x;
    if (i < total) {
        int node = i >> 6;
        float dv = dinv[node];
        agg[i] = dv * dv * h[i];
    }
}

// one wave per edge, 2 features per lane; native f32 atomics into L2-resident agg
__global__ __launch_bounds__(256) void k_edges(
    const int* __restrict__ src, const int* __restrict__ dst,
    const float* __restrict__ dinv, const float* __restrict__ h,
    float* __restrict__ agg, int E) {
    const int lane = threadIdx.x & 31;
    int w  = (blockIdx.x * blockDim.x + threadIdx.x) >> 5;
    int nw = (gridDim.x * blockDim.x) >> 5;
    for (int e = w; e < E; e += nw) {
        int s = src[e], d = dst[e];
        float nm = dinv[s] * dinv[d];
        const float2 hv = *(const float2*)(h + ((size_t)s << 6) + (lane << 1));
        float* ap = agg + ((size_t)d << 6) + (lane << 1);
        unsafeAtomicAdd(ap,     nm * hv.x);
        unsafeAtomicAdd(ap + 1, nm * hv.y);
    }
}

// ---------------- combine ----------------
__global__ void k_combine(const float* __restrict__ agg, const float* __restrict__ r,
                          float* __restrict__ out, int total) {
    int i = blockIdx.x * blockDim.x + threadIdx.x;
    if (i < total) out[i] = fmaxf(agg[i], 0.0f) + r[i];
}

__global__ void k_combine_last(const float* __restrict__ agg, const float* __restrict__ r,
                               float* __restrict__ out, int total) {  // total = N*40
    int i = blockIdx.x * blockDim.x + threadIdx.x;
    if (i < total) {
        int node = i / NCLS, c = i - node * NCLS;
        int j = (node << 6) + c;                    // padded 64-wide buffers
        out[i] = fmaxf(agg[j], 0.0f) + r[j];
    }
}

static inline size_t alignup(size_t v) { return (v + 255) & ~(size_t)255; }

extern "C" void kernel_launch(void* const* d_in, const int* in_sizes, int n_in,
                              void* d_out, int out_size, void* d_ws, size_t ws_size,
                              hipStream_t stream) {
    (void)n_in; (void)out_size; (void)ws_size;
    const float* x   = (const float*)d_in[0];
    const int*   ei  = (const int*)d_in[1];
    const int    E   = in_sizes[1] / 2;
    const int*   src = ei;
    const int*   dst = ei + E;
    const float* W[4] = {(const float*)d_in[2], (const float*)d_in[3],
                         (const float*)d_in[4], (const float*)d_in[5]};
    const float* R[4] = {(const float*)d_in[6], (const float*)d_in[7],
                         (const float*)d_in[8], (const float*)d_in[9]};

    // scratch layout
    char* p = (char*)d_ws;
    float* dinv = (float*)p;  p += alignup((size_t)NNODES * 4);
    float* h    = (float*)p;  p += alignup((size_t)NNODES * FDIM * 4);
    float* r    = (float*)p;  p += alignup((size_t)NNODES * FDIM * 4);
    float* agg  = (float*)p;  p += alignup((size_t)NNODES * FDIM * 4);
    float* xbuf = (float*)p;  p += alignup((size_t)NNODES * FDIM * 4);
    __bf16* xb  = (__bf16*)p; p += alignup((size_t)NNODES * FDIM * 2);
    __bf16* wbT = (__bf16*)p; p += alignup((size_t)64 * 64 * 2);
    __bf16* rbT = (__bf16*)p; p += alignup((size_t)64 * 64 * 2);

    const int NF = NNODES * FDIM;

    // symmetric normalization: deg (with self-loop) -> D^-1/2
    k_init_deg <<<(NNODES + 255) / 256, 256, 0, stream>>>(dinv, NNODES);
    k_count_deg<<<(E + 255) / 256,      256, 0, stream>>>(dst, E, dinv);
    k_dinv     <<<(NNODES + 255) / 256, 256, 0, stream>>>(dinv, NNODES);

    const float* xin = x;
    for (int l = 0; l < 4; ++l) {
        int ncols = (l == 3) ? NCLS : FDIM;
        k_cvt_x<<<(NF / 2 + 255) / 256, 256, 0, stream>>>(
            (const float2*)xin, (unsigned int*)xb, NF / 2);
        k_cvt_w<<<16, 256, 0, stream>>>(W[l], R[l], wbT, rbT, ncols);
        k_gemm_dual<<<3125, 256, 0, stream>>>(xb, wbT, rbT, h, r);   // 25000 waves
        k_self <<<(NF + 255) / 256, 256, 0, stream>>>(dinv, h, agg, NF);
        k_edges<<<(E + 7) / 8, 256, 0, stream>>>(src, dst, dinv, h, agg, E);
        if (l < 3) {
            k_combine<<<(NF + 255) / 256, 256, 0, stream>>>(agg, r, xbuf, NF);
            xin = xbuf;
        } else {
            k_combine_last<<<(NNODES * NCLS + 255) / 256, 256, 0, stream>>>(
                agg, r, (float*)d_out, NNODES * NCLS);
        }
    }
}
